// DefaultDecoder_75788992906359
// MI455X (gfx1250) — matrix operand order repaired and checked by reading, not compiled, not run on hardware
//
#include <hip/hip_runtime.h>
#include <hip/hip_bf16.h>
#include <stdint.h>

#define T_STEPS 28
#define BATCH   2048
#define LATD    256
#define DHID    1024
#define E_DIM   256
#define LN_EPS  1e-5f

typedef __attribute__((ext_vector_type(16))) __bf16 v16bf;
typedef __attribute__((ext_vector_type(8)))  __bf16 v8bf;
typedef __attribute__((ext_vector_type(8)))  float  v8f;

__device__ __forceinline__ unsigned short f32_to_bf16_bits(float f) {
    uint32_t u = __float_as_uint(f);
    u += 0x7FFFu + ((u >> 16) & 1u);   // round-to-nearest-even
    return (unsigned short)(u >> 16);
}

// ---------------------------------------------------------------- conversions
__global__ __launch_bounds__(256)
void cvt_f32_to_bf16(const float* __restrict__ src, unsigned short* __restrict__ dst, int n) {
    int i = blockIdx.x * blockDim.x + threadIdx.x;
    int stride = gridDim.x * blockDim.x;
    for (; i < n; i += stride) dst[i] = f32_to_bf16_bits(src[i]);
}

// ---------------------------------------------------------------- WMMA fragment loads
// A fragment (16x32 bf16, MxK), ISA 7.12.2: lanes 0-15 row M=lane, K=0..7 (v0-3) & 16..23 (v4-7);
// lanes 16-31 row M=lane-16, K=8..15 & 24..31.
__device__ __forceinline__ v16bf load_a_frag(const __bf16* __restrict__ A, int lda,
                                             int m0, int k0, int lane) {
    const int row = m0 + (lane & 15);
    const int kb  = k0 + ((lane >> 4) << 3);
    v8bf lo = *(const v8bf*)(A + (size_t)row * lda + kb);
    v8bf hi = *(const v8bf*)(A + (size_t)row * lda + kb + 16);
    v16bf a;
#pragma unroll
    for (int i = 0; i < 8; ++i) { a[i] = lo[i]; a[i + 8] = hi[i]; }
    return a;
}

// B fragment (32x16 bf16, KxN) from W stored row-major [N_total x K] (i.e. we compute A·Wᵀ):
// lane n = output column; lanes 0-15 hold K=k0..k0+7 & k0+16..k0+23, lanes 16-31 hold K=k0+8..k0+15 & k0+24..k0+31.
__device__ __forceinline__ v16bf load_b_frag(const __bf16* __restrict__ W, int ldw,
                                             int n0, int k0, int lane) {
    const int col = n0 + (lane & 15);
    const int kb  = k0 + ((lane >> 4) << 3);
    v8bf lo = *(const v8bf*)(W + (size_t)col * ldw + kb);
    v8bf hi = *(const v8bf*)(W + (size_t)col * ldw + kb + 16);
    v16bf b;
#pragma unroll
    for (int i = 0; i < 8; ++i) { b[i] = lo[i]; b[i + 8] = hi[i]; }
    return b;
}

__device__ __forceinline__ v8f wmma_bf16(v16bf a, v16bf b, v8f c) {
    return __builtin_amdgcn_wmma_f32_16x16x32_bf16(false, a, false, b, (short)0, c,
                                                   false, false);
}

// ---------------------------------------------------------------- generic GEMM: C = A·Wᵀ + bias
// Block = 256 threads = 8 waves (2 along M x 4 along N). Wave tile 32x64 -> block tile 64x256.
__global__ __launch_bounds__(256)
void gemm_bf16_wmma(const unsigned short* __restrict__ Abits,
                    const unsigned short* __restrict__ Wbits,
                    const float* __restrict__ bias, float* __restrict__ C,
                    int M, int N, int K) {
    const __bf16* A = (const __bf16*)Abits;
    const __bf16* W = (const __bf16*)Wbits;
    const int lane = threadIdx.x & 31;
    const int wave = threadIdx.x >> 5;
    const int m0 = blockIdx.x * 64  + (wave & 1) * 32;
    const int n0 = blockIdx.y * 256 + (wave >> 1) * 64;

    const v8f zero = {0.f, 0.f, 0.f, 0.f, 0.f, 0.f, 0.f, 0.f};
    v8f c[2][4];
#pragma unroll
    for (int mi = 0; mi < 2; ++mi)
#pragma unroll
        for (int j = 0; j < 4; ++j) c[mi][j] = zero;

    for (int k0 = 0; k0 < K; k0 += 32) {
        v16bf a0 = load_a_frag(A, K, m0,      k0, lane);
        v16bf a1 = load_a_frag(A, K, m0 + 16, k0, lane);
#pragma unroll
        for (int j = 0; j < 4; ++j) {
            v16bf b = load_b_frag(W, K, n0 + 16 * j, k0, lane);
            c[0][j] = wmma_bf16(a0, b, c[0][j]);
            c[1][j] = wmma_bf16(a1, b, c[1][j]);
        }
    }

    // C/D layout: VGPR i -> row m0 + (lane<16 ? i : 8+i), col n0 + (lane&15)
    const int rbase = m0 + ((lane >> 4) << 3);
    const int cbase = n0 + (lane & 15);
#pragma unroll
    for (int mi = 0; mi < 2; ++mi)
#pragma unroll
        for (int j = 0; j < 4; ++j) {
            const int col = cbase + 16 * j;
            const float bv = bias[col];
#pragma unroll
            for (int i = 0; i < 8; ++i) {
                const int row = rbase + 16 * mi + i;
                C[(size_t)row * N + col] = c[mi][j][i] + bv;
            }
        }
}

// ---------------------------------------------------------------- fused GRU step
// gh = h_in · w_hhᵀ + b_hh never leaves registers: each wave accumulates the r/z/n
// column tiles for the SAME output columns, then the epilogue applies the gate math.
// Wave tile 32x32 per gate (12 v8f accumulators). Block = 8 waves -> 64x128 h-tile.
__global__ __launch_bounds__(256)
void gru_step_fused(const unsigned short* __restrict__ hinbits,  // [B,D] bf16 (A)
                    const unsigned short* __restrict__ whhbits,  // [3D,D] bf16
                    const float* __restrict__ b_hh,              // [3D]
                    const float* __restrict__ gi,                // [B,3D]
                    const float* __restrict__ hin,               // [B,D] f32
                    float* __restrict__ h,                       // [B,D] next state
                    float* __restrict__ out_t) {                 // [B,D] output slice
    const __bf16* A = (const __bf16*)hinbits;
    const __bf16* W = (const __bf16*)whhbits;
    const int lane = threadIdx.x & 31;
    const int wave = threadIdx.x >> 5;
    const int m0 = blockIdx.x * 64  + (wave & 1) * 32;
    const int n0 = blockIdx.y * 128 + (wave >> 1) * 32;

    const v8f zero = {0.f, 0.f, 0.f, 0.f, 0.f, 0.f, 0.f, 0.f};
    v8f acc[3][2][2];   // [gate r/z/n][mi][nj]
#pragma unroll
    for (int g = 0; g < 3; ++g)
#pragma unroll
        for (int mi = 0; mi < 2; ++mi)
#pragma unroll
            for (int nj = 0; nj < 2; ++nj) acc[g][mi][nj] = zero;

    for (int k0 = 0; k0 < DHID; k0 += 32) {
        v16bf a0 = load_a_frag(A, DHID, m0,      k0, lane);
        v16bf a1 = load_a_frag(A, DHID, m0 + 16, k0, lane);
#pragma unroll
        for (int g = 0; g < 3; ++g) {
            const int ng = n0 + g * DHID;
#pragma unroll
            for (int nj = 0; nj < 2; ++nj) {
                v16bf b = load_b_frag(W, DHID, ng + 16 * nj, k0, lane);
                acc[g][0][nj] = wmma_bf16(a0, b, acc[g][0][nj]);
                acc[g][1][nj] = wmma_bf16(a1, b, acc[g][1][nj]);
            }
        }
    }

    // epilogue: gate math in registers, write h_new only
    const int rbase = m0 + ((lane >> 4) << 3);
    const int cb    = lane & 15;
#pragma unroll
    for (int mi = 0; mi < 2; ++mi)
#pragma unroll
        for (int nj = 0; nj < 2; ++nj) {
            const int col = n0 + 16 * nj + cb;
            const float br = b_hh[col];
            const float bz = b_hh[col + DHID];
            const float bn = b_hh[col + 2 * DHID];
#pragma unroll
            for (int i = 0; i < 8; ++i) {
                const int row = rbase + 16 * mi + i;
                const size_t gbase = (size_t)row * (3 * DHID) + col;
                const float ghr = acc[0][mi][nj][i] + br;
                const float ghz = acc[1][mi][nj][i] + bz;
                const float ghn = acc[2][mi][nj][i] + bn;
                const float r = 1.f / (1.f + expf(-(gi[gbase] + ghr)));
                const float u = 1.f / (1.f + expf(-(gi[gbase + DHID] + ghz)));
                const float n = tanhf(gi[gbase + 2 * DHID] + r * ghn);
                const float hi = hin[(size_t)row * DHID + col];
                const float hnew = (1.f - u) * n + u * hi;
                h[(size_t)row * DHID + col]     = hnew;
                out_t[(size_t)row * DHID + col] = hnew;
            }
        }
}

// ---------------------------------------------------------------- LayerNorm (one row per block)
__global__ __launch_bounds__(256)
void ln_kernel(const float* __restrict__ xin, const float* __restrict__ lnw,
               const float* __restrict__ lnb, float* __restrict__ h0,
               unsigned short* __restrict__ xbf) {
    __shared__ float s_sum[256], s_sq[256];
    const int row = blockIdx.x;
    const float* xr = xin + (size_t)row * DHID;
    float v[4], lsum = 0.f, lsq = 0.f;
#pragma unroll
    for (int j = 0; j < 4; ++j) {
        v[j] = xr[threadIdx.x + 256 * j];
        lsum += v[j]; lsq += v[j] * v[j];
    }
    s_sum[threadIdx.x] = lsum; s_sq[threadIdx.x] = lsq;
    __syncthreads();
    for (int off = 128; off > 0; off >>= 1) {
        if (threadIdx.x < off) {
            s_sum[threadIdx.x] += s_sum[threadIdx.x + off];
            s_sq[threadIdx.x]  += s_sq[threadIdx.x + off];
        }
        __syncthreads();
    }
    const float mu  = s_sum[0] * (1.f / DHID);
    const float var = s_sq[0] * (1.f / DHID) - mu * mu;
    const float inv = rsqrtf(var + LN_EPS);
#pragma unroll
    for (int j = 0; j < 4; ++j) {
        const int d = threadIdx.x + 256 * j;
        const float y = (v[j] - mu) * inv * lnw[d] + lnb[d];
        h0[(size_t)row * DHID + d]  = y;
        xbf[(size_t)row * DHID + d] = f32_to_bf16_bits(y);
    }
}

// ---------------------------------------------------------------- teacher-forcing select
__global__ __launch_bounds__(256)
void select_kernel(const float* __restrict__ h, const float* __restrict__ true_inp,
                   const int* __restrict__ mask, int t,
                   float* __restrict__ hin, unsigned short* __restrict__ hin_bf) {
    const size_t idx = (size_t)blockIdx.x * blockDim.x + threadIdx.x;
    const int b = (int)(idx >> 10), d = (int)(idx & 1023);
    const bool m = mask[t] != 0;
    const float v = m ? true_inp[((size_t)b * T_STEPS + t) * E_DIM + (d & (E_DIM - 1))]
                      : h[idx];
    hin[idx]    = v;
    hin_bf[idx] = f32_to_bf16_bits(v);
}

// ---------------------------------------------------------------- host launcher
extern "C" void kernel_launch(void* const* d_in, const int* in_sizes, int n_in,
                              void* d_out, int out_size, void* d_ws, size_t ws_size,
                              hipStream_t stream) {
    (void)in_sizes; (void)n_in; (void)out_size; (void)ws_size;
    const float* z        = (const float*)d_in[0];
    const float* true_inp = (const float*)d_in[1];
    const int*   tf_mask  = (const int*)  d_in[2];
    const float* fc_w     = (const float*)d_in[3];
    const float* fc_b     = (const float*)d_in[4];
    const float* ln_w     = (const float*)d_in[5];
    const float* ln_b     = (const float*)d_in[6];
    const float* w_ih     = (const float*)d_in[7];
    const float* b_ih     = (const float*)d_in[8];
    const float* w_hh     = (const float*)d_in[9];
    const float* b_hh     = (const float*)d_in[10];
    float* out = (float*)d_out;

    char* ws = (char*)d_ws;
    size_t off = 0;
    auto wsalloc = [&](size_t bytes) -> void* {
        void* p = ws + off;
        off += (bytes + 255) & ~(size_t)255;
        return p;
    };
    unsigned short* z_bf   = (unsigned short*)wsalloc((size_t)BATCH * LATD * 2);
    unsigned short* fcw_bf = (unsigned short*)wsalloc((size_t)DHID * LATD * 2);
    unsigned short* wih_bf = (unsigned short*)wsalloc((size_t)3 * DHID * DHID * 2);
    unsigned short* whh_bf = (unsigned short*)wsalloc((size_t)3 * DHID * DHID * 2);
    float*          xpre   = (float*)         wsalloc((size_t)BATCH * DHID * 4);
    unsigned short* x_bf   = (unsigned short*)wsalloc((size_t)BATCH * DHID * 2);
    float*          h      = (float*)         wsalloc((size_t)BATCH * DHID * 4);
    float*          hin    = (float*)         wsalloc((size_t)BATCH * DHID * 4);
    unsigned short* hin_bf = (unsigned short*)wsalloc((size_t)BATCH * DHID * 2);
    float*          gi     = (float*)         wsalloc((size_t)BATCH * 3 * DHID * 4);

    // 1) one-time bf16 conversions (weights stay L2-resident afterwards)
    cvt_f32_to_bf16<<<1024, 256, 0, stream>>>(z,    z_bf,   BATCH * LATD);
    cvt_f32_to_bf16<<<1024, 256, 0, stream>>>(fc_w, fcw_bf, DHID * LATD);
    cvt_f32_to_bf16<<<4096, 256, 0, stream>>>(w_ih, wih_bf, 3 * DHID * DHID);
    cvt_f32_to_bf16<<<4096, 256, 0, stream>>>(w_hh, whh_bf, 3 * DHID * DHID);

    // 2) fc GEMM + LayerNorm -> h0 (f32) and x (bf16)
    gemm_bf16_wmma<<<dim3(BATCH / 64, DHID / 256), 256, 0, stream>>>(
        z_bf, fcw_bf, fc_b, xpre, BATCH, DHID, LATD);
    ln_kernel<<<BATCH, 256, 0, stream>>>(xpre, ln_w, ln_b, h, x_bf);

    // 3) input gates, computed once: gi = x · w_ihᵀ + b_ih
    gemm_bf16_wmma<<<dim3(BATCH / 64, (3 * DHID) / 256), 256, 0, stream>>>(
        x_bf, wih_bf, b_ih, gi, BATCH, 3 * DHID, DHID);

    // 4) 28 sequential GRU steps; recurrent GEMM + gates fused in one kernel
    const int n_elem = BATCH * DHID;
    for (int t = 0; t < T_STEPS; ++t) {
        select_kernel<<<n_elem / 256, 256, 0, stream>>>(h, true_inp, tf_mask, t, hin, hin_bf);
        gru_step_fused<<<dim3(BATCH / 64, DHID / 128), 256, 0, stream>>>(
            hin_bf, whh_bf, b_hh, gi, hin, h, out + (size_t)t * n_elem);
    }
}
